// DiagnosticNet_58497454571623
// MI455X (gfx1250) — matrix operand order repaired and checked by reading, not compiled, not run on hardware
//
#include <hip/hip_runtime.h>
#include <hip/hip_bf16.h>

// ---- Problem constants (match reference) ----
#define BATCH 16384
#define DIMX  2048
#define KCLS  128

#define ROWS     64     // rows per block (4 M-tiles)
#define NTILES   8      // KCLS / 16
#define KSTEPS   64     // DIMX / 32
#define FRAG_DW  256    // dwords per fragment: 32 lanes * 16 bf16 / 2
#define FRAGBUF_DW (NTILES * KSTEPS * FRAG_DW)   // 131072 dwords = 512 KB
#define SS_STRIDE 132   // padded score-row stride (avoid 64-bank conflicts)

// ---- CDNA5 WMMA vector types ----
typedef __bf16        bf16x16 __attribute__((ext_vector_type(16)));
typedef unsigned int  u32x8   __attribute__((ext_vector_type(8)));
typedef float         f32x8   __attribute__((ext_vector_type(8)));

// pack high-16 bits (bf16 truncation) of two floats into one dword: f0 -> [15:0], f1 -> [31:16]
__device__ __forceinline__ unsigned pack_bf16_pair(float f0, float f1) {
    return __builtin_amdgcn_perm(__float_as_uint(f1), __float_as_uint(f0), 0x07060302u);
}
__device__ __forceinline__ float trunc_bf16_f(float f) {
    return __uint_as_float(__float_as_uint(f) & 0xFFFF0000u);
}

__device__ __forceinline__ f32x8 wmma_bf16(bf16x16 a, bf16x16 b, f32x8 c) {
    // 8 args: (neg_a, A, neg_b, B, c_mod, C, reuse_a, reuse_b)
    return __builtin_amdgcn_wmma_f32_16x16x32_bf16(false, a, false, b,
                                                   (short)0, c, false, false);
}

__device__ __forceinline__ bf16x16 frag_from_2x(uint4 a, uint4 b) {
    u32x8 p;
    p[0] = a.x; p[1] = a.y; p[2] = a.z; p[3] = a.w;
    p[4] = b.x; p[5] = b.y; p[6] = b.z; p[7] = b.w;
    return __builtin_bit_cast(bf16x16, p);
}

// ================= prep: split enc_W into bf16 hi/lo WMMA B-fragments =================
// Fragment (nt, ks): B-matrix 32(k) x 16(n). Lane L: col n = nt*16 + (L&15),
// k = ks*32 + (L>>4)*16 + e, e = 0..15 contiguous. Stored fragment-major, 32B/lane.
__global__ __launch_bounds__(256)
void prep_encW_kernel(const float* __restrict__ enc_W,
                      unsigned* __restrict__ frag_hi,
                      unsigned* __restrict__ frag_lo)
{
    int t    = blockIdx.x * 256 + threadIdx.x;   // 0 .. NTILES*KSTEPS*32-1
    int lane = t & 31;
    int frag = t >> 5;                           // nt*KSTEPS + ks
    int ks   = frag & (KSTEPS - 1);
    int nt   = frag >> 6;
    int n    = nt * 16 + (lane & 15);
    int kb   = ks * 32 + (lane >> 4) * 16;
    const float* w = enc_W + (size_t)n * DIMX + kb;

    unsigned hi[8], lo[8];
#pragma unroll
    for (int i = 0; i < 8; ++i) {
        float f0 = w[2 * i], f1 = w[2 * i + 1];
        hi[i] = pack_bf16_pair(f0, f1);
        lo[i] = pack_bf16_pair(f0 - trunc_bf16_f(f0), f1 - trunc_bf16_f(f1));
    }
    unsigned* dh = frag_hi + (size_t)frag * FRAG_DW + lane * 8;
    unsigned* dl = frag_lo + (size_t)frag * FRAG_DW + lane * 8;
    *(uint4*)(dh)     = make_uint4(hi[0], hi[1], hi[2], hi[3]);
    *(uint4*)(dh + 4) = make_uint4(hi[4], hi[5], hi[6], hi[7]);
    *(uint4*)(dl)     = make_uint4(lo[0], lo[1], lo[2], lo[3]);
    *(uint4*)(dl + 4) = make_uint4(lo[4], lo[5], lo[6], lo[7]);
}

// ================= fused: GEMM -> argmax -> gathered decoder dot =================
// One block = 64 rows of x, 128 threads = 4 wave32.
// Wave w owns class tiles (2w, 2w+1) for ALL 4 M-tiles -> each B fragment feeds 12 WMMAs.
__global__ __launch_bounds__(128)
void fused_route_kernel(const float* __restrict__ x,
                        const float* __restrict__ y,
                        const unsigned* __restrict__ frag_hi,
                        const unsigned* __restrict__ frag_lo,
                        const float* __restrict__ enc_b,
                        const float* __restrict__ dec_W,
                        const float* __restrict__ dec_bias,
                        float* __restrict__ out_yhat,
                        float* __restrict__ out_zidx,
                        float* __restrict__ loss_part,
                        float* __restrict__ acc_part)
{
    // x chunk staged as packed bf16 hi/lo: ROWS x 64 k -> ROWS x 32 dwords
    __shared__ __align__(16) unsigned xs_hi[ROWS * 32];   // 8 KB
    __shared__ __align__(16) unsigned xs_lo[ROWS * 32];   // 8 KB
    __shared__ float ss[ROWS * SS_STRIDE];                // ~33 KB
    __shared__ float pmax[ROWS][2];
    __shared__ int   pidx[ROWS][2];
    __shared__ int   zidx[ROWS];
    __shared__ float lpart[4], apart[4];

    const int tid  = threadIdx.x;
    const int lane = tid & 31;
    const int wid  = tid >> 5;
    const int row0 = blockIdx.x * ROWS;

    const int m16 = lane & 15;
    const int g   = lane >> 4;
    const int nt0 = wid * 2;          // class tiles
    const int nt1 = nt0 + 1;

    f32x8 acc[4][2];
#pragma unroll
    for (int mt = 0; mt < 4; ++mt) { acc[mt][0] = (f32x8){}; acc[mt][1] = (f32x8){}; }

    for (int kb = 0; kb < DIMX; kb += 64) {
        // ---- stage + split x chunk (cooperative; 8 consecutive floats/thread/iter) ----
#pragma unroll
        for (int it = 0; it < 4; ++it) {
            int lin = tid + it * 128;          // 0..511
            int r   = lin >> 3;                // row 0..63
            int c0  = (lin & 7) * 8;           // col 0,8,..,56
            const float* src = x + (size_t)(row0 + r) * DIMX + kb + c0;
            float4 v0 = ((const float4*)src)[0];
            float4 v1 = ((const float4*)src)[1];
            if (kb + 64 < DIMX)
                __builtin_prefetch(src + 64, 0, 3);   // global_prefetch_b8

            uint4 h4 = make_uint4(pack_bf16_pair(v0.x, v0.y),
                                  pack_bf16_pair(v0.z, v0.w),
                                  pack_bf16_pair(v1.x, v1.y),
                                  pack_bf16_pair(v1.z, v1.w));
            float l0 = v0.x - trunc_bf16_f(v0.x), l1 = v0.y - trunc_bf16_f(v0.y);
            float l2 = v0.z - trunc_bf16_f(v0.z), l3 = v0.w - trunc_bf16_f(v0.w);
            float l4 = v1.x - trunc_bf16_f(v1.x), l5 = v1.y - trunc_bf16_f(v1.y);
            float l6 = v1.z - trunc_bf16_f(v1.z), l7 = v1.w - trunc_bf16_f(v1.w);
            uint4 lo4 = make_uint4(pack_bf16_pair(l0, l1), pack_bf16_pair(l2, l3),
                                   pack_bf16_pair(l4, l5), pack_bf16_pair(l6, l7));
            *(uint4*)(xs_hi + r * 32 + c0 / 2) = h4;
            *(uint4*)(xs_lo + r * 32 + c0 / 2) = lo4;
        }
        __syncthreads();

#pragma unroll
        for (int kc = 0; kc < 64; kc += 32) {
            // ---- B fragments: precomputed, fragment-major, fully coalesced ----
            const int ks = (kb + kc) >> 5;
            const unsigned* bp0h = frag_hi + (size_t)(nt0 * KSTEPS + ks) * FRAG_DW + lane * 8;
            const unsigned* bp0l = frag_lo + (size_t)(nt0 * KSTEPS + ks) * FRAG_DW + lane * 8;
            const unsigned* bp1h = frag_hi + (size_t)(nt1 * KSTEPS + ks) * FRAG_DW + lane * 8;
            const unsigned* bp1l = frag_lo + (size_t)(nt1 * KSTEPS + ks) * FRAG_DW + lane * 8;
            bf16x16 b0h = frag_from_2x(*(const uint4*)(bp0h), *(const uint4*)(bp0h + 4));
            bf16x16 b0l = frag_from_2x(*(const uint4*)(bp0l), *(const uint4*)(bp0l + 4));
            bf16x16 b1h = frag_from_2x(*(const uint4*)(bp1h), *(const uint4*)(bp1h + 4));
            bf16x16 b1l = frag_from_2x(*(const uint4*)(bp1l), *(const uint4*)(bp1l + 4));

            const int o0 = (kc + g * 16) >> 1;       // dword offset, run k=g*16+e
            const int o1 = (kc + g * 16 + 8) >> 1;   // run k=g*16+8+e

            // ---- sweep 4 M-tiles with these B fragments ----
#pragma unroll
            for (int mt = 0; mt < 4; ++mt) {
                const unsigned* xr_hi = xs_hi + (mt * 16 + m16) * 32;
                const unsigned* xr_lo = xs_lo + (mt * 16 + m16) * 32;
                bf16x16 a_hi = frag_from_2x(*(const uint4*)(xr_hi + o0),
                                            *(const uint4*)(xr_hi + o1));
                bf16x16 a_lo = frag_from_2x(*(const uint4*)(xr_lo + o0),
                                            *(const uint4*)(xr_lo + o1));
                acc[mt][0] = wmma_bf16(a_hi, b0h, acc[mt][0]);
                acc[mt][0] = wmma_bf16(a_hi, b0l, acc[mt][0]);
                acc[mt][0] = wmma_bf16(a_lo, b0h, acc[mt][0]);
                acc[mt][1] = wmma_bf16(a_hi, b1h, acc[mt][1]);
                acc[mt][1] = wmma_bf16(a_hi, b1l, acc[mt][1]);
                acc[mt][1] = wmma_bf16(a_lo, b1h, acc[mt][1]);
            }
        }
        __syncthreads();
    }

    // ---------------- epilogue: bias + scores to LDS ----------------
    {
        const int n0 = nt0 * 16, n1 = nt1 * 16;
        float bn0 = enc_b[n0 + m16];
        float bn1 = enc_b[n1 + m16];
#pragma unroll
        for (int mt = 0; mt < 4; ++mt) {
#pragma unroll
            for (int j = 0; j < 8; ++j) {
                int m = mt * 16 + j + g * 8;       // C layout: VGPR j, lane half g
                ss[m * SS_STRIDE + n0 + m16] = acc[mt][0][j] + bn0;
                ss[m * SS_STRIDE + n1 + m16] = acc[mt][1][j] + bn1;
            }
        }
    }
    __syncthreads();

    // ---------------- argmax over K=128 (first-max tie-break) ----------------
    {
        int m = tid >> 1, p = tid & 1;             // 2 threads per row
        float best = -3.402823466e+38f;
        int bi = 0;
#pragma unroll 8
        for (int q = 0; q < 64; ++q) {
            int n = p * 64 + q;
            float v = ss[m * SS_STRIDE + n];
            if (v > best) { best = v; bi = n; }
        }
        pmax[m][p] = best;
        pidx[m][p] = bi;
    }
    __syncthreads();
    if (tid < ROWS) {
        float b0 = pmax[tid][0];
        int   i0 = pidx[tid][0];
        zidx[tid] = (pmax[tid][1] > b0) ? pidx[tid][1] : i0;
    }
    __syncthreads();

    // ------------- decoder dot (per-wave, wave32 shuffle reduce; x rows L2-hot) -------------
    {
        float l_acc = 0.f, a_acc = 0.f;
#pragma unroll 2
        for (int m = 0; m < 16; ++m) {
            const int r = wid * 16 + m;
            const float* xrow = x + (size_t)(row0 + r) * DIMX;
            const float* wrow = dec_W + (size_t)zidx[r] * DIMX;
            float s = 0.f;
#pragma unroll 8
            for (int i = lane; i < DIMX; i += 32)
                s = fmaf(xrow[i], wrow[i], s);
#pragma unroll
            for (int off = 16; off > 0; off >>= 1)
                s += __shfl_down(s, off, 32);
            if (lane == 0) {
                int rg = row0 + r;
                float yv = s + dec_bias[zidx[r]];
                out_yhat[rg] = yv;
                out_zidx[rg] = (float)zidx[r];
                float yt = y[rg];
                float d = yv - yt;
                float sg = (yv > 0.f) ? 1.f : ((yv < 0.f) ? -1.f : 0.f);
                l_acc += d * d;
                a_acc += (sg == yt) ? 1.f : 0.f;
            }
        }
        if (lane == 0) { lpart[wid] = l_acc; apart[wid] = a_acc; }
    }
    __syncthreads();
    if (tid == 0) {
        loss_part[blockIdx.x] = lpart[0] + lpart[1] + lpart[2] + lpart[3];
        acc_part[blockIdx.x]  = apart[0] + apart[1] + apart[2] + apart[3];
    }
}

// ================= deterministic final reduction =================
__global__ __launch_bounds__(256)
void reduce_kernel(const float* __restrict__ loss_part,
                   const float* __restrict__ acc_part,
                   float* __restrict__ out2,
                   int nblocks, float invB)
{
    __shared__ float sl[256], sa[256];
    int tid = threadIdx.x;
    float l = 0.f, a = 0.f;
    for (int i = tid; i < nblocks; i += 256) { l += loss_part[i]; a += acc_part[i]; }
    sl[tid] = l; sa[tid] = a;
    __syncthreads();
#pragma unroll
    for (int st = 128; st > 0; st >>= 1) {
        if (tid < st) { sl[tid] += sl[tid + st]; sa[tid] += sa[tid + st]; }
        __syncthreads();
    }
    if (tid == 0) { out2[0] = sl[0] * invB; out2[1] = sa[0] * invB; }
}

extern "C" void kernel_launch(void* const* d_in, const int* in_sizes, int n_in,
                              void* d_out, int out_size, void* d_ws, size_t ws_size,
                              hipStream_t stream) {
    const float* x        = (const float*)d_in[0];
    const float* y        = (const float*)d_in[1];
    // d_in[2] = z (unused by reference outputs)
    const float* enc_W    = (const float*)d_in[3];
    const float* enc_b    = (const float*)d_in[4];
    const float* dec_W    = (const float*)d_in[5];   // [1,K,D] flat
    const float* dec_bias = (const float*)d_in[6];

    float* out = (float*)d_out;                      // [y_hat(B), z_idx(B), loss, acc]
    unsigned* wsu = (unsigned*)d_ws;
    unsigned* frag_hi = wsu;                         // 512 KB
    unsigned* frag_lo = wsu + FRAGBUF_DW;            // 512 KB
    float* loss_part  = (float*)(wsu + 2 * FRAGBUF_DW);
    float* acc_part   = loss_part + (BATCH / ROWS);

    const int nblocks = BATCH / ROWS;                // 256

    prep_encW_kernel<<<(NTILES * KSTEPS * 32) / 256, 256, 0, stream>>>(
        enc_W, frag_hi, frag_lo);

    fused_route_kernel<<<nblocks, 128, 0, stream>>>(
        x, y, frag_hi, frag_lo, enc_b, dec_W, dec_bias,
        out, out + BATCH, loss_part, acc_part);

    reduce_kernel<<<1, 256, 0, stream>>>(loss_part, acc_part,
                                         out + 2 * BATCH, nblocks,
                                         1.0f / (float)BATCH);
}
